// AttLearner_10969346474295
// MI455X (gfx1250) — matrix operand order repaired and checked by reading, not compiled, not run on hardware
//
#include <hip/hip_runtime.h>
#include <hip/hip_bf16.h>
#include <stdint.h>

#define NROW 10000
#define DIM  512
#define NP   10112      // padded rows: 79*128
#define KSEL 31         // top-(K+1) kept per row
#define EPS  1e-12f

#define AST 40          // A LDS row stride (bf16): 80B -> conflict-free frag reads
#define BST 136         // B LDS row stride (bf16): 272B -> conflict-free frag reads
#define NS  (DIM / 32)  // 16 k-steps

typedef __attribute__((ext_vector_type(16))) __bf16 v16bf;
typedef __attribute__((ext_vector_type(8)))  float  v8f;

// ---- bf16 helpers (explicit RNE via integer ops) ----
__device__ __forceinline__ __bf16 f2bf(float f) {
    unsigned u = __float_as_uint(f);
    unsigned r = (u + 0x7FFFu + ((u >> 16) & 1u)) >> 16;   // round-nearest-even
    unsigned short s = (unsigned short)r;
    return __builtin_bit_cast(__bf16, s);
}
__device__ __forceinline__ float bf2f(__bf16 b) {
    unsigned short s = __builtin_bit_cast(unsigned short, b);
    return __uint_as_float(((unsigned)s) << 16);
}

// load a 32-byte WMMA fragment from two 16-byte LDS chunks
__device__ __forceinline__ v16bf ld_frag(const __bf16* p0, const __bf16* p1) {
    union { v16bf v; uint4 q[2]; } u;
    u.q[0] = *reinterpret_cast<const uint4*>(p0);
    u.q[1] = *reinterpret_cast<const uint4*>(p1);
    return u.v;
}

// async global -> LDS copy of 16 bytes per lane (CDNA5 async path, ASYNCcnt)
__device__ __forceinline__ void async_b128_to_lds(const void* gptr, void* lptr) {
    unsigned ldsOff = (unsigned)(uintptr_t)lptr;   // generic LDS addr low 32 bits
    asm volatile("global_load_async_to_lds_b128 %0, %1, off"
                 :: "v"(ldsOff), "v"(gptr) : "memory");
}
__device__ __forceinline__ void wait_async0() {
    asm volatile("s_wait_asynccnt 0" ::: "memory");
}

// =====================================================================
// Kernel 1: h = (relu(x*w0))*w1 ; row-normalize ; split into bf16 hi+lo
// One block (256 thr) per padded row; padded rows written as zeros.
// =====================================================================
__global__ void __launch_bounds__(256)
k_norm_split(const float* __restrict__ x, const float* __restrict__ w,
             __bf16* __restrict__ hi, __bf16* __restrict__ lo) {
    const int row = blockIdx.x;
    const int t   = threadIdx.x;
    const int k0  = t, k1 = t + 256;
    __shared__ float red[256];

    float h0 = 0.f, h1 = 0.f;
    if (row < NROW) {
        float a = x[(size_t)row * DIM + k0] * w[k0];
        a = a > 0.f ? a : 0.f;  a *= w[DIM + k0];
        float b = x[(size_t)row * DIM + k1] * w[k1];
        b = b > 0.f ? b : 0.f;  b *= w[DIM + k1];
        h0 = a; h1 = b;
    }
    red[t] = h0 * h0 + h1 * h1;
    __syncthreads();
    for (int s = 128; s > 0; s >>= 1) {
        if (t < s) red[t] += red[t + s];
        __syncthreads();
    }
    const float rs = rsqrtf(red[0] + EPS);
    const float n0 = h0 * rs, n1 = h1 * rs;
    const __bf16 h0h = f2bf(n0), h1h = f2bf(n1);
    hi[(size_t)row * DIM + k0] = h0h;
    hi[(size_t)row * DIM + k1] = h1h;
    lo[(size_t)row * DIM + k0] = f2bf(n0 - bf2f(h0h));
    lo[(size_t)row * DIM + k1] = f2bf(n1 - bf2f(h1h));
}

// =====================================================================
// Kernel 2: adj = hn * hn^T via bf16x2 split WMMA (hi*hi + hi*lo + lo*hi)
// 128x128 block tile, 512 thr = 16 waves, 2x2 16x16 acc/wave.
// Software-pipelined: 2 LDS stages; async A staging + B scatter for step
// ks+1 overlap the 12 WMMAs of step ks. One barrier per k-step.
// =====================================================================
#define A_ELEMS  (2 * 2 * 128 * AST)           // [stage][hi/lo][m][k]
#define B_ELEMS  (2 * 2 * 32  * BST)           // [stage][hi/lo][k][n]
#define SMEM_BYTES ((A_ELEMS + B_ELEMS) * 2)

__global__ void __launch_bounds__(512)
k_gemm(const __bf16* __restrict__ hi, const __bf16* __restrict__ lo,
       float* __restrict__ out) {
    extern __shared__ __align__(16) __bf16 smem[];
    __bf16* sAs = smem;                        // A_ELEMS
    __bf16* sBs = smem + A_ELEMS;              // B_ELEMS
#define AS(s, b, m, k) sAs[((((s)*2 + (b)) * 128) + (m)) * AST + (k)]
#define BS(s, b, kk, n) sBs[((((s)*2 + (b)) * 32) + (kk)) * BST + (n)]

    const int t     = threadIdx.x;
    const int mBase = blockIdx.y * 128;
    const int nBase = blockIdx.x * 128;

    const int lane = t & 31;
    const int wave = t >> 5;                // 0..15
    const int wm   = (wave & 3) * 32;       // wave M offset in block tile
    const int wn   = (wave >> 2) * 32;      // wave N offset in block tile
    const int lrow = lane & 15;
    const int lhk  = lane >> 4;             // K-half select for A frags

    v8f acc00 = {}, acc01 = {}, acc10 = {}, acc11 = {};

    // staging index maps (512 threads)
    const int arow = t >> 2;                // 0..127
    const int akc  = (t & 3) * 8;           // 0,8,16,24
    const int bn   = t >> 2;                // 0..127
    const int bkc  = (t & 3) * 8;           // 0,8,16,24

    const size_t aOff = (size_t)(mBase + arow) * DIM + akc;
    const size_t bOff = (size_t)(nBase + bn)   * DIM + bkc;

    auto stageA = [&](int kk, int st) {
        async_b128_to_lds(&hi[aOff + kk], &AS(st, 0, arow, akc));
        async_b128_to_lds(&lo[aOff + kk], &AS(st, 1, arow, akc));
    };
    auto scatterB = [&](uint4 q, int st, int b) {
        union { uint4 q; __bf16 e[8]; } u; u.q = q;
        #pragma unroll
        for (int i = 0; i < 8; ++i) BS(st, b, bkc + i, bn) = u.e[i];
    };

    // ---- prologue: stage k-step 0 into stage 0 ----
    stageA(0, 0);
    uint4 bh = *(const uint4*)&hi[bOff];
    uint4 bl = *(const uint4*)&lo[bOff];
    scatterB(bh, 0, 0);
    scatterB(bl, 0, 1);

    int cur = 0;
    for (int ks = 0; ks < NS; ++ks) {
        wait_async0();            // async A writes for stage `cur` complete
        __syncthreads();          // stage `cur` visible to all waves

        const bool nxt = (ks + 1 < NS);
        if (nxt) {                // issue next stage early (latency hiding)
            const int kk2 = (ks + 1) * 32;
            stageA(kk2, cur ^ 1);
            bh = *(const uint4*)&hi[bOff + kk2];
            bl = *(const uint4*)&lo[bOff + kk2];
            if (ks + 2 < NS)      // keep HBM ahead on first pass
                __builtin_prefetch(&hi[aOff + kk2 + 32], 0, 1);
        }

        // A fragments: lane<16 holds row m, K{0..7,16..23}; lane>=16 K{8..15,24..31}
        v16bf a0h = ld_frag(&AS(cur,0, wm +      lrow, lhk*8), &AS(cur,0, wm +      lrow, 16 + lhk*8));
        v16bf a1h = ld_frag(&AS(cur,0, wm + 16 + lrow, lhk*8), &AS(cur,0, wm + 16 + lrow, 16 + lhk*8));
        v16bf a0l = ld_frag(&AS(cur,1, wm +      lrow, lhk*8), &AS(cur,1, wm +      lrow, 16 + lhk*8));
        v16bf a1l = ld_frag(&AS(cur,1, wm + 16 + lrow, lhk*8), &AS(cur,1, wm + 16 + lrow, 16 + lhk*8));
        // B fragments: lane%16 = N; lane<16 K{0..7,16..23}, lane>=16 K{8..15,24..31}
        v16bf b0h, b1h, b0l, b1l;
        #pragma unroll
        for (int s = 0; s < 16; ++s) {
            const int kb = (s < 8) ? (lhk*8 + s) : (16 + lhk*8 + (s - 8));
            b0h[s] = BS(cur,0, kb, wn + lrow);
            b1h[s] = BS(cur,0, kb, wn + 16 + lrow);
            b0l[s] = BS(cur,1, kb, wn + lrow);
            b1l[s] = BS(cur,1, kb, wn + 16 + lrow);
        }

        // D = A*B + C ; bf16x2: hi*hi + hi*lo + lo*hi (lo*lo ~1e-5, dropped)
        acc00 = __builtin_amdgcn_wmma_f32_16x16x32_bf16(false, a0h, false, b0h, (short)0, acc00, false, false);
        acc00 = __builtin_amdgcn_wmma_f32_16x16x32_bf16(false, a0h, false, b0l, (short)0, acc00, false, false);
        acc00 = __builtin_amdgcn_wmma_f32_16x16x32_bf16(false, a0l, false, b0h, (short)0, acc00, false, false);

        acc01 = __builtin_amdgcn_wmma_f32_16x16x32_bf16(false, a0h, false, b1h, (short)0, acc01, false, false);
        acc01 = __builtin_amdgcn_wmma_f32_16x16x32_bf16(false, a0h, false, b1l, (short)0, acc01, false, false);
        acc01 = __builtin_amdgcn_wmma_f32_16x16x32_bf16(false, a0l, false, b1h, (short)0, acc01, false, false);

        acc10 = __builtin_amdgcn_wmma_f32_16x16x32_bf16(false, a1h, false, b0h, (short)0, acc10, false, false);
        acc10 = __builtin_amdgcn_wmma_f32_16x16x32_bf16(false, a1h, false, b0l, (short)0, acc10, false, false);
        acc10 = __builtin_amdgcn_wmma_f32_16x16x32_bf16(false, a1l, false, b0h, (short)0, acc10, false, false);

        acc11 = __builtin_amdgcn_wmma_f32_16x16x32_bf16(false, a1h, false, b1h, (short)0, acc11, false, false);
        acc11 = __builtin_amdgcn_wmma_f32_16x16x32_bf16(false, a1h, false, b1l, (short)0, acc11, false, false);
        acc11 = __builtin_amdgcn_wmma_f32_16x16x32_bf16(false, a1l, false, b1h, (short)0, acc11, false, false);

        if (nxt) {                // scatter B for next stage after the WMMAs
            scatterB(bh, cur ^ 1, 0);
            scatterB(bl, cur ^ 1, 1);
        }
        cur ^= 1;
    }

    // C/D layout: VGPR e -> M = e + 8*(lane>=16); N = lane%16
    const int r0 = mBase + wm + lhk * 8;
    const int c0 = nBase + wn + lrow;
    {
        v8f accs[4] = { acc00, acc01, acc10, acc11 };
        #pragma unroll
        for (int tm = 0; tm < 2; ++tm) {
            #pragma unroll
            for (int tn = 0; tn < 2; ++tn) {
                v8f a = accs[tm * 2 + tn];
                const int col = c0 + tn * 16;
                if (col < NROW) {
                    #pragma unroll
                    for (int e = 0; e < 8; ++e) {
                        const int r = r0 + tm * 16 + e;
                        if (r < NROW) out[(size_t)r * NROW + col] = a[e];
                    }
                }
            }
        }
    }
#undef AS
#undef BS
}

// =====================================================================
// Kernel 3: per-row exact radix-select of the 31st-largest value, then
// in-place threshold + relu. One block (256 thr) per row.
// =====================================================================
__device__ __forceinline__ unsigned fkey(float f) {
    unsigned u = __float_as_uint(f);
    return (u & 0x80000000u) ? ~u : (u | 0x80000000u);   // order-preserving
}

__global__ void __launch_bounds__(256)
k_select(float* __restrict__ out) {
    const int row = blockIdx.x;
    float* rp = out + (size_t)row * NROW;
    const int t = threadIdx.x;

    __shared__ unsigned hist[256];
    __shared__ unsigned sPrefix, sNeed;
    if (t == 0) { sPrefix = 0u; sNeed = KSEL; }
    __syncthreads();

    for (int pass = 0; pass < 4; ++pass) {
        hist[t] = 0u;
        __syncthreads();
        const unsigned pfx = sPrefix;
        const int shift = 24 - pass * 8;
        for (int j = t; j < NROW; j += 256) {
            const unsigned k = fkey(rp[j]);
            if (pass == 0 || (k >> (shift + 8)) == pfx)
                atomicAdd(&hist[(k >> shift) & 255u], 1u);
        }
        __syncthreads();
        if (t == 0) {
            unsigned need = sNeed, c = 0u, b = 0u;
            for (int bb = 255; bb >= 0; --bb) {
                c += hist[bb];
                if (c >= need) { b = (unsigned)bb; need -= (c - hist[bb]); break; }
            }
            sPrefix = (sPrefix << 8) | b;
            sNeed = need;
        }
        __syncthreads();
    }

    const unsigned T = sPrefix;   // exact key of the 31st-largest value
    for (int j = t; j < NROW; j += 256) {
        const float f = rp[j];
        rp[j] = (fkey(f) >= T && f > 0.f) ? f : 0.f;
    }
}

// =====================================================================
extern "C" void kernel_launch(void* const* d_in, const int* in_sizes, int n_in,
                              void* d_out, int out_size, void* d_ws, size_t ws_size,
                              hipStream_t stream) {
    const float* x = (const float*)d_in[0];   // [10000, 512]
    const float* w = (const float*)d_in[1];   // [2, 512]
    float* out = (float*)d_out;               // [10000, 10000]

    __bf16* hi = (__bf16*)d_ws;               // [NP, 512] bf16 high part
    __bf16* lo = hi + (size_t)NP * DIM;       // [NP, 512] bf16 residual

    k_norm_split<<<NP, 256, 0, stream>>>(x, w, hi, lo);

    dim3 grid(NP / 128, NP / 128, 1);         // (79, 79)
    k_gemm<<<grid, 512, SMEM_BYTES, stream>>>(hi, lo, out);

    k_select<<<NROW, 256, 0, stream>>>(out);
}
